// TransformerBlock_25666724560978
// MI455X (gfx1250) — matrix operand order repaired and checked by reading, compiled, not run on hardware
//
#include <hip/hip_runtime.h>
#include <math.h>

// ---------------- problem constants ----------------
#define B_  2
#define S_  512
#define D_  512
#define H_  8
#define DH_ 64
#define F_  2048
#define P_  16
#define T_  (B_ * S_)   // 1024 tokens

typedef __attribute__((ext_vector_type(16))) __bf16 v16bf;
typedef __attribute__((ext_vector_type(8)))  float  v8f;

// fp32 -> bf16 round-to-nearest-even (bit pattern)
__device__ __forceinline__ unsigned short f2bf(float f) {
    unsigned int u = __float_as_uint(f);
    u += 0x7fffu + ((u >> 16) & 1u);
    return (unsigned short)(u >> 16);
}

union Frag16 { uint4 q[2]; v16bf v; };

// A fragment (16x32 bf16, MxK): lane 0-15 row M=lane, K = {0..7,16..23};
// lane 16-31 same rows, K = {8..15,24..31}.  base = row start + kk.
__device__ __forceinline__ v16bf load_a_frag(const unsigned short* base, int half) {
    Frag16 f;
    f.q[0] = *reinterpret_cast<const uint4*>(base + half * 8);
    f.q[1] = *reinterpret_cast<const uint4*>(base + 16 + half * 8);
    return f.v;
}

// B fragment (32x16 bf16, KxN), sourced from Bt[N][K] (column-contig-in-K):
// lanes 0-15 hold K = {0..7,16..23}, lanes 16-31 hold K = {8..15,24..31}. base = Bt row(n) + kk.
__device__ __forceinline__ v16bf load_b_frag(const unsigned short* base, int half) {
    Frag16 f;
    f.q[0] = *reinterpret_cast<const uint4*>(base + half * 8);
    f.q[1] = *reinterpret_cast<const uint4*>(base + 16 + half * 8);
    return f.v;
}

__device__ __forceinline__ v8f wmma_bf16(v16bf a, v16bf b, v8f c) {
    return __builtin_amdgcn_wmma_f32_16x16x32_bf16(false, a, false, b, (short)0, c, false, false);
}

// ---------------------------------------------------------------------------
// Generic bf16 WMMA GEMM: C[M,N] = A[M,K] * B[K,N] (+ Cin), B given as
// Bt[N][K] row-major bf16. One wave per block; wave computes a 32x64 tile
// (2 M-subtiles x 4 N-subtiles, 8 WMMAs per k-step, B frags reused 2x).
// Requires M%32==0, N%64==0, K%32==0.
// ---------------------------------------------------------------------------
__global__ void gemm_bf16_wmma(const unsigned short* __restrict__ A,
                               const unsigned short* __restrict__ Bt,
                               float* __restrict__ C,
                               const float* __restrict__ Cin,
                               int M, int N, int Kd) {
    const int lane = threadIdx.x;
    const int half = lane >> 4;
    const int lr   = lane & 15;
    const int n0 = blockIdx.x * 64;
    const int m0 = blockIdx.y * 32;

    v8f acc[2][4];
    if (Cin) {
#pragma unroll
        for (int mt = 0; mt < 2; ++mt)
#pragma unroll
            for (int nt = 0; nt < 4; ++nt)
#pragma unroll
                for (int r = 0; r < 8; ++r)
                    acc[mt][nt][r] =
                        Cin[(size_t)(m0 + mt * 16 + half * 8 + r) * N + n0 + nt * 16 + lr];
    } else {
#pragma unroll
        for (int mt = 0; mt < 2; ++mt)
#pragma unroll
            for (int nt = 0; nt < 4; ++nt)
#pragma unroll
                for (int r = 0; r < 8; ++r) acc[mt][nt][r] = 0.0f;
    }

    const unsigned short* arow0 = A + (size_t)(m0 + lr) * Kd;
    const unsigned short* arow1 = A + (size_t)(m0 + 16 + lr) * Kd;
    for (int kk = 0; kk < Kd; kk += 32) {
        v16bf af0 = load_a_frag(arow0 + kk, half);
        v16bf af1 = load_a_frag(arow1 + kk, half);
        __builtin_prefetch(arow0 + kk + 64, 0, 0);   // global_prefetch_b8
#pragma unroll
        for (int nt = 0; nt < 4; ++nt) {
            const unsigned short* brow = Bt + (size_t)(n0 + nt * 16 + lr) * Kd + kk;
            v16bf bf_ = load_b_frag(brow, half);
            acc[0][nt] = wmma_bf16(af0, bf_, acc[0][nt]);
            acc[1][nt] = wmma_bf16(af1, bf_, acc[1][nt]);
        }
    }

#pragma unroll
    for (int mt = 0; mt < 2; ++mt)
#pragma unroll
        for (int nt = 0; nt < 4; ++nt)
#pragma unroll
            for (int r = 0; r < 8; ++r)
                C[(size_t)(m0 + mt * 16 + half * 8 + r) * N + n0 + nt * 16 + lr] =
                    acc[mt][nt][r];
}

// ---------------------------------------------------------------------------
// fp32 [R,C] row-major -> bf16 [C,R] row-major (transpose + convert), batched.
// block (32,8), grid (C/32, R/32, batch)
// ---------------------------------------------------------------------------
__global__ void transpose_cvt(const float* __restrict__ in,
                              unsigned short* __restrict__ out, int R, int C) {
    __shared__ float tile[32][33];
    const size_t mb = (size_t)blockIdx.z * R * C;
    const int c0 = blockIdx.x * 32, r0 = blockIdx.y * 32;
#pragma unroll
    for (int j = 0; j < 32; j += 8)
        tile[threadIdx.y + j][threadIdx.x] =
            in[mb + (size_t)(r0 + threadIdx.y + j) * C + c0 + threadIdx.x];
    __syncthreads();
#pragma unroll
    for (int j = 0; j < 32; j += 8)
        out[mb + (size_t)(c0 + threadIdx.y + j) * R + r0 + threadIdx.x] =
            f2bf(tile[threadIdx.x][threadIdx.y + j]);
}

// ---------------------------------------------------------------------------
// RMSNorm row kernel (D=512), one block (256 thr) per token row.
// ---------------------------------------------------------------------------
__global__ void rmsnorm_cast(const float* __restrict__ x, const float* __restrict__ g,
                             unsigned short* __restrict__ obf, float* __restrict__ of) {
    const int t = blockIdx.x;
    const int i = threadIdx.x;
    const float* xr = x + (size_t)t * D_;
    float v0 = xr[i], v1 = xr[i + 256];
    __shared__ float red[256];
    red[i] = v0 * v0 + v1 * v1;
    __syncthreads();
    for (int s = 128; s; s >>= 1) {
        if (i < s) red[i] += red[i + s];
        __syncthreads();
    }
    float scale = rsqrtf(red[0] / (float)D_ + 1e-6f);
    float y0 = v0 * scale * g[i];
    float y1 = v1 * scale * g[i + 256];
    obf[(size_t)t * D_ + i] = f2bf(y0);
    obf[(size_t)t * D_ + i + 256] = f2bf(y1);
    if (of) {
        of[(size_t)t * D_ + i] = y0;
        of[(size_t)t * D_ + i + 256] = y1;
    }
}

// ---------------------------------------------------------------------------
// Repack fp32 [T, H*DH] -> bf16 [B,H,S,DH] (tr==0) or [B,H,DH,S] (tr==1)
// ---------------------------------------------------------------------------
__global__ void pack_qkv(const float* __restrict__ src,
                         unsigned short* __restrict__ dst, int tr) {
    const int idx = blockIdx.x * blockDim.x + threadIdx.x; // over T_*512
    const int t = idx >> 9, c = idx & 511;
    const int h = c >> 6, dh = c & 63;
    const int b = t >> 9, s = t & 511;
    unsigned short v = f2bf(src[idx]);
    const size_t bh = (size_t)(b * H_ + h);
    if (!tr) dst[(bh * S_ + s) * DH_ + dh] = v;
    else     dst[(bh * DH_ + dh) * S_ + s] = v;
}

// ---------------------------------------------------------------------------
// Causal attention for one (b, h, 16-query block). Block = 128 thr (4 waves).
// Phase 0: async-copy Q tile (16x64 bf16 = 2KB) into LDS (ASYNCcnt path)
// Phase 1: WMMA Q*K^T -> LDS scores (masked, scaled), A-frags from LDS
// Phase 2: parallel exact softmax in LDS (8 lanes per row)
// Phase 3: WMMA probs*V, write bf16 [B,S,H*DH]
// ---------------------------------------------------------------------------
__global__ void attention_kernel(const unsigned short* __restrict__ Qb,
                                 const unsigned short* __restrict__ Kb,
                                 const unsigned short* __restrict__ Vt,
                                 unsigned short* __restrict__ attnO) {
    __shared__ __align__(16) float          sc[16][S_];
    __shared__ __align__(16) unsigned short pb[16][S_];
    __shared__ __align__(16) unsigned short qs[16][DH_];

    const int b = blockIdx.z, h = blockIdx.y, q0 = blockIdx.x * 16;
    const int wave = threadIdx.x >> 5;
    const int lane = threadIdx.x & 31;
    const int half = lane >> 4, lr = lane & 15;
    const size_t bh = (size_t)(b * H_ + h);

    // phase 0: async global->LDS staging of the Q tile (one b128 per lane)
    {
        const unsigned short* gsrc = Qb + (bh * S_ + q0) * DH_ + (size_t)threadIdx.x * 8;
        unsigned lds_off = (unsigned)(size_t)(&qs[0][0] + threadIdx.x * 8);
        asm volatile("global_load_async_to_lds_b128 %0, %1, off"
                     :: "v"(lds_off), "v"(gsrc) : "memory");
        asm volatile("s_wait_asynccnt 0x0" ::: "memory");
    }
    __syncthreads();

    const unsigned short* Kbase = Kb + bh * S_ * DH_;

    // phase 1: score tiles (A from LDS)
    for (int kt = wave; kt < S_ / 16; kt += 4) {
        v8f acc = {};
        const unsigned short* brow = Kbase + (size_t)(kt * 16 + lr) * DH_;
#pragma unroll
        for (int kk = 0; kk < DH_; kk += 32) {
            v16bf af  = load_a_frag(&qs[lr][kk], half);    // ds_load_b128
            v16bf bf_ = load_b_frag(brow + kk, half);
            acc = wmma_bf16(af, bf_, acc);
        }
        const int key = kt * 16 + lr;
#pragma unroll
        for (int r = 0; r < 8; ++r) {
            const int row = half * 8 + r;
            const int q = q0 + row;
            sc[row][key] = (key <= q) ? acc[r] * 0.125f : -3.0e38f;
        }
    }
    __syncthreads();

    // phase 2: softmax, 8 lanes per query row (groups are 8-aligned in a wave)
    {
        const int row = threadIdx.x >> 3;
        const int sub = threadIdx.x & 7;
        const int k0 = sub * (S_ / 8);
        float m = -3.4e38f;
        for (int k = k0; k < k0 + S_ / 8; ++k) m = fmaxf(m, sc[row][k]);
        for (int msk = 4; msk; msk >>= 1) m = fmaxf(m, __shfl_xor(m, msk, 32));
        float s = 0.f;
        for (int k = k0; k < k0 + S_ / 8; ++k) {
            float e = __expf(sc[row][k] - m);
            sc[row][k] = e;
            s += e;
        }
        for (int msk = 4; msk; msk >>= 1) s += __shfl_xor(s, msk, 32);
        const float inv = 1.0f / s;
        for (int k = k0; k < k0 + S_ / 8; ++k) pb[row][k] = f2bf(sc[row][k] * inv);
    }
    __syncthreads();

    // phase 3: out tile for dh columns [16*wave, 16*wave+16)
    const int dh0 = wave * 16;
    v8f acc = {};
    const unsigned short* Vbase = Vt + bh * DH_ * S_ + (size_t)(dh0 + lr) * S_;
    for (int kk = 0; kk < S_; kk += 32) {
        v16bf af  = load_a_frag(&pb[lr][kk], half);   // ds_load_b128 from LDS
        v16bf bf_ = load_b_frag(Vbase + kk, half);
        acc = wmma_bf16(af, bf_, acc);
    }
#pragma unroll
    for (int r = 0; r < 8; ++r) {
        const int q = q0 + half * 8 + r;
        attnO[((size_t)b * S_ + q) * (H_ * DH_) + h * DH_ + dh0 + lr] = f2bf(acc[r]);
    }
}

// ---------------------------------------------------------------------------
// Routing: softmax over P=16 primitives, top-4, renormalize -> dense gate[T,16]
// One wave per token; lanes 0-15 own one primitive each (16-31 mirror).
// ---------------------------------------------------------------------------
__global__ void routing_topk(const float* __restrict__ act,
                             const float* __restrict__ Rm,
                             float* __restrict__ gate, int dim) {
    const int wave = threadIdx.x >> 5;
    const int lane = threadIdx.x & 31;
    const int t = blockIdx.x * 8 + wave;
    const int p = lane & 15;

    const float* a = act + (size_t)t * dim;
    const float* r = Rm + (size_t)p * dim;
    float dot = 0.f;
    for (int d = 0; d < dim; ++d) dot += a[d] * r[d];

    float m = dot;
    for (int msk = 8; msk; msk >>= 1) m = fmaxf(m, __shfl_xor(m, msk, 32));
    float e = __expf(dot - m);
    float s = e;
    for (int msk = 8; msk; msk >>= 1) s += __shfl_xor(s, msk, 32);
    const float prob = e / s;

    float v = prob;
    bool sel = false;
#pragma unroll
    for (int it = 0; it < 4; ++it) {
        float bv = v; int bi = p;
        for (int msk = 8; msk; msk >>= 1) {
            float ov = __shfl_xor(bv, msk, 32);
            int   oi = __shfl_xor(bi, msk, 32);
            if (ov > bv || (ov == bv && oi < bi)) { bv = ov; bi = oi; }
        }
        if (p == bi) { sel = true; v = -1.0f; }
    }
    float ssum = sel ? prob : 0.f;
    for (int msk = 8; msk; msk >>= 1) ssum += __shfl_xor(ssum, msk, 32);
    if (lane < 16) gate[(size_t)t * P_ + p] = sel ? prob / ssum : 0.f;
}

// A_p[t][d] = bf16( act[t][d] * gate[t][p] )   (gate is 0 for unselected)
__global__ void scale_cast(const float* __restrict__ act, const float* __restrict__ gate,
                           int p, int dim, unsigned short* __restrict__ out, int total) {
    const int idx = blockIdx.x * blockDim.x + threadIdx.x;
    if (idx >= total) return;
    const int t = idx / dim;
    out[idx] = f2bf(act[idx] * gate[t * P_ + p]);
}

__global__ void gelu_k(const float* __restrict__ in, float* __restrict__ out, int total) {
    const int idx = blockIdx.x * blockDim.x + threadIdx.x;
    if (idx >= total) return;
    const float v = in[idx];
    out[idx] = 0.5f * v * (1.0f + erff(v * 0.70710678118654752f));
}

__global__ void copy_f32(const float* __restrict__ in, float* __restrict__ out, int n) {
    const int i = blockIdx.x * blockDim.x + threadIdx.x;
    if (i < n) out[i] = in[i];
}

// ---------------------------------------------------------------------------
extern "C" void kernel_launch(void* const* d_in, const int* in_sizes, int n_in,
                              void* d_out, int out_size, void* d_ws, size_t ws_size,
                              hipStream_t stream) {
    (void)in_sizes; (void)n_in; (void)out_size; (void)ws_size;
    const float* x  = (const float*)d_in[0];
    const float* g1 = (const float*)d_in[1];
    const float* g2 = (const float*)d_in[2];
    const float* Wq = (const float*)d_in[3];
    const float* Wk = (const float*)d_in[4];
    const float* Wv = (const float*)d_in[5];
    const float* Wo = (const float*)d_in[6];
    const float* R1 = (const float*)d_in[7];
    const float* W1 = (const float*)d_in[8];
    const float* R2 = (const float*)d_in[9];
    const float* W2 = (const float*)d_in[10];
    float* out = (float*)d_out;

    // ---- workspace bump allocator ----
    char* w = (char*)d_ws;
    auto alloc = [&](size_t bytes) {
        char* p = w;
        w += (bytes + 255) & ~(size_t)255;
        return p;
    };
    unsigned short* WqT  = (unsigned short*)alloc((size_t)D_ * D_ * 2);
    unsigned short* WkT  = (unsigned short*)alloc((size_t)D_ * D_ * 2);
    unsigned short* WvT  = (unsigned short*)alloc((size_t)D_ * D_ * 2);
    unsigned short* WoT  = (unsigned short*)alloc((size_t)D_ * D_ * 2);
    unsigned short* W1T  = (unsigned short*)alloc((size_t)P_ * D_ * F_ * 2);
    unsigned short* W2T  = (unsigned short*)alloc((size_t)P_ * F_ * D_ * 2);
    unsigned short* h1bf = (unsigned short*)alloc((size_t)T_ * D_ * 2);
    float* qf = (float*)alloc((size_t)T_ * D_ * 4);
    float* kf = (float*)alloc((size_t)T_ * D_ * 4);
    float* vf = (float*)alloc((size_t)T_ * D_ * 4);
    unsigned short* Qb = (unsigned short*)alloc((size_t)T_ * D_ * 2);
    unsigned short* Kb = (unsigned short*)alloc((size_t)T_ * D_ * 2);
    unsigned short* Vt = (unsigned short*)alloc((size_t)T_ * D_ * 2);
    unsigned short* attnbf = (unsigned short*)alloc((size_t)T_ * D_ * 2);
    float* x1  = (float*)alloc((size_t)T_ * D_ * 4);
    unsigned short* h2bf = (unsigned short*)alloc((size_t)T_ * D_ * 2);
    float* h2f = (float*)alloc((size_t)T_ * D_ * 4);
    float* gate1 = (float*)alloc((size_t)T_ * P_ * 4);
    float* gate2 = (float*)alloc((size_t)T_ * P_ * 4);
    unsigned short* Asc = (unsigned short*)alloc((size_t)T_ * F_ * 2); // reused (max dim)
    float* hmid = (float*)alloc((size_t)T_ * F_ * 4);
    float* af   = (float*)alloc((size_t)T_ * F_ * 4);

    const dim3 tb(32, 8);

    // 1) convert + transpose all weights to bf16 Bt layout
    transpose_cvt<<<dim3(D_ / 32, D_ / 32, 1), tb, 0, stream>>>(Wq, WqT, D_, D_);
    transpose_cvt<<<dim3(D_ / 32, D_ / 32, 1), tb, 0, stream>>>(Wk, WkT, D_, D_);
    transpose_cvt<<<dim3(D_ / 32, D_ / 32, 1), tb, 0, stream>>>(Wv, WvT, D_, D_);
    transpose_cvt<<<dim3(D_ / 32, D_ / 32, 1), tb, 0, stream>>>(Wo, WoT, D_, D_);
    transpose_cvt<<<dim3(F_ / 32, D_ / 32, P_), tb, 0, stream>>>(W1, W1T, D_, F_);
    transpose_cvt<<<dim3(D_ / 32, F_ / 32, P_), tb, 0, stream>>>(W2, W2T, F_, D_);

    // 2) h1 = rmsnorm(x, g1)  (bf16)
    rmsnorm_cast<<<T_, 256, 0, stream>>>(x, g1, h1bf, nullptr);

    // 3) Q/K/V projections (WMMA)
    gemm_bf16_wmma<<<dim3(D_ / 64, T_ / 32), 32, 0, stream>>>(h1bf, WqT, qf, nullptr, T_, D_, D_);
    gemm_bf16_wmma<<<dim3(D_ / 64, T_ / 32), 32, 0, stream>>>(h1bf, WkT, kf, nullptr, T_, D_, D_);
    gemm_bf16_wmma<<<dim3(D_ / 64, T_ / 32), 32, 0, stream>>>(h1bf, WvT, vf, nullptr, T_, D_, D_);

    // 4) repack to attention layouts
    const int tot = T_ * D_;
    pack_qkv<<<tot / 256, 256, 0, stream>>>(qf, Qb, 0);
    pack_qkv<<<tot / 256, 256, 0, stream>>>(kf, Kb, 0);
    pack_qkv<<<tot / 256, 256, 0, stream>>>(vf, Vt, 1);

    // 5) causal attention (WMMA + async-to-LDS staging)
    attention_kernel<<<dim3(S_ / 16, H_, B_), 128, 0, stream>>>(Qb, Kb, Vt, attnbf);

    // 6) output projection + residual: x1 = attn @ Wo + x
    gemm_bf16_wmma<<<dim3(D_ / 64, T_ / 32), 32, 0, stream>>>(attnbf, WoT, x1, x, T_, D_, D_);

    // 7) h2 = rmsnorm(x1, g2)
    rmsnorm_cast<<<T_, 256, 0, stream>>>(x1, g2, h2bf, h2f);

    // 8) fc1 routing (softmax -> top4 -> renorm, dense gate)
    routing_topk<<<T_ / 8, 256, 0, stream>>>(h2f, R1, gate1, D_);

    // 9) expert bank 1: hmid = sum_p (gate1[:,p] * h2) @ W1[p]
    for (int p = 0; p < P_; ++p) {
        scale_cast<<<tot / 256, 256, 0, stream>>>(h2f, gate1, p, D_, Asc, tot);
        gemm_bf16_wmma<<<dim3(F_ / 64, T_ / 32), 32, 0, stream>>>(
            Asc, W1T + (size_t)p * F_ * D_, hmid, p ? hmid : nullptr, T_, F_, D_);
    }

    // 10) exact GELU
    const int totF = T_ * F_;
    gelu_k<<<totF / 256, 256, 0, stream>>>(hmid, af, totF);

    // 11) fc2 routing
    routing_topk<<<T_ / 8, 256, 0, stream>>>(af, R2, gate2, F_);

    // 12) out = x1 + sum_p (gate2[:,p] * a) @ W2[p]
    copy_f32<<<tot / 256, 256, 0, stream>>>(x1, out, tot);
    for (int p = 0; p < P_; ++p) {
        scale_cast<<<totF / 256, 256, 0, stream>>>(af, gate2, p, F_, Asc, totF);
        gemm_bf16_wmma<<<dim3(D_ / 64, T_ / 32), 32, 0, stream>>>(
            Asc, W2T + (size_t)p * D_ * F_, out, out, T_, D_, F_);
    }
}
